// UserModelPW_22308060136186
// MI455X (gfx1250) — matrix operand order repaired and checked by reading, not compiled, not run on hardware
//
#include <hip/hip_runtime.h>
#include <hip/hip_bf16.h>
#include <math.h>

// ---------------------------------------------------------------------------
// CDNA5 (gfx1250) fused implementation.
// Fixed model dims from the reference: F=64, H=256. Dynamic: S, ND, band, per, PW.
// ---------------------------------------------------------------------------

typedef __attribute__((ext_vector_type(16))) _Float16 v16h;
typedef __attribute__((ext_vector_type(8)))  _Float16 v8h;
typedef __attribute__((ext_vector_type(8)))  float    v8f;

#define FDIM 64
#define HDIM 256
#define K1DIM 128              // hist(64) + disp_feat(64)
#define ROWS_PER_WAVE 16
#define WAVES_PER_BLOCK 4
#define ROWS_PER_BLOCK (ROWS_PER_WAVE * WAVES_PER_BLOCK)   // 64

// Branchless ELU using the hardware transcendental (v_exp_f32, TRANS pipe —
// co-executes with the WMMA/XDL pipe instead of a 25-op libm polynomial).
__device__ __forceinline__ float elu_fast(float x) {
    return fmaxf(x, 0.f) + (__expf(fminf(x, 0.f)) - 1.f);
}

// ---------------------------------------------------------------------------
// K0: build f16 transposed weight blocks in workspace.
//   bw1t[n][k] (k<64: sum of the PW identical history blocks of W1; k>=64: disp block)
//   w2t[n][k]  = W2[k][n]
// ---------------------------------------------------------------------------
__global__ void prep_weights_kernel(const float* __restrict__ W1,
                                    const float* __restrict__ W2,
                                    _Float16* __restrict__ bw1t,
                                    _Float16* __restrict__ w2t,
                                    int PW) {
    int idx = blockIdx.x * blockDim.x + threadIdx.x;
    if (idx < HDIM * K1DIM) {
        int n = idx >> 7;           // 0..255
        int k = idx & 127;          // 0..127
        float v;
        if (k < FDIM) {
            v = 0.f;
            for (int p = 0; p < PW; ++p)
                v += W1[(size_t)(p * FDIM + k) * HDIM + n];
        } else {
            v = W1[(size_t)(PW * FDIM + (k - FDIM)) * HDIM + n];
        }
        bw1t[(size_t)n * K1DIM + k] = (_Float16)v;
    } else {
        int j = idx - HDIM * K1DIM;
        if (j < HDIM * HDIM) {
            int n = j >> 8;
            int k = j & 255;
            w2t[(size_t)n * HDIM + k] = (_Float16)W2[(size_t)k * HDIM + n];
        }
    }
}

// ---------------------------------------------------------------------------
// K1: banded history sum.  hist16[s][f] = 1e-4 * sum_j Xs[tril_col(s,j)][f]
// ---------------------------------------------------------------------------
__global__ void hist_kernel(const float* __restrict__ Xs,
                            const int* __restrict__ tril,   // [NT,2] flattened
                            _Float16* __restrict__ hist16,
                            int band) {
    int s = blockIdx.x;
    int f = threadIdx.x;                 // blockDim.x == 64
    long long base = (long long)s * band;
    float acc = 0.f;
    for (int j = 0; j < band; ++j) {
        int col = tril[2 * (base + j) + 1];
        acc += Xs[(size_t)col * FDIM + f];
    }
    hist16[(size_t)s * FDIM + f] = (_Float16)(1e-4f * acc);
}

// ---------------------------------------------------------------------------
// Fragment helpers (layouts per CDNA5 ISA §7.12.2, wave32):
//  A 16-bit 16x32: lane<16 -> row M=lane, K = {b..b+7, b+16..b+23} with b=0
//                  lane>=16 -> same rows, b=8
//  B 16-bit 32x16: lane<16 -> col N=lane, K = {b..b+7, b+16..b+23} with b=0 ; lane>=16 -> b=8
// ---------------------------------------------------------------------------
__device__ __forceinline__ v16h load_afrag(const _Float16* row, int ks, int lane) {
    int kb = ks * 32 + ((lane >> 4) << 3);        // +0 or +8
    v8h lo = *(const v8h*)(row + kb);
    v8h hi = *(const v8h*)(row + kb + 16);
    return __builtin_shufflevector(lo, hi, 0,1,2,3,4,5,6,7,8,9,10,11,12,13,14,15);
}

__device__ __forceinline__ v16h load_bfrag(const _Float16* __restrict__ Bt,
                                           int rowlen, int n0, int ks, int lane) {
    const _Float16* p = Bt + (size_t)(n0 + (lane & 15)) * rowlen
                           + ks * 32 + ((lane >> 4) << 3);   // +0 or +8
    v8h lo = *(const v8h*)p;
    v8h hi = *(const v8h*)(p + 16);
    return __builtin_shufflevector(lo, hi, 0,1,2,3,4,5,6,7,8,9,10,11,12,13,14,15);
}

// ---------------------------------------------------------------------------
// K2: fused MLP. Each wave32 owns a 16-row tile.
//  layer1: A=[hist|dispfeat] (K=128) x bw1t -> elu -> Hs (LDS, f16)
//  layer2: Hs (K=256) x w2t -> elu -> dot with W3 (shfl_xor reduce) -> u, exp(u)
// ---------------------------------------------------------------------------
__global__ void __launch_bounds__(128)
mlp_fused_kernel(const float* __restrict__ dispf,
                 const int* __restrict__ dispsec,
                 const _Float16* __restrict__ hist16,
                 const _Float16* __restrict__ bw1t,
                 const _Float16* __restrict__ w2t,
                 const float* __restrict__ b1,
                 const float* __restrict__ b2,
                 const float* __restrict__ W3,
                 const float* __restrict__ b3p,
                 float* __restrict__ u_out,
                 float* __restrict__ expu_out,
                 int ND) {
    __shared__ __align__(16) _Float16 As[ROWS_PER_BLOCK][K1DIM];   // 16 KB
    __shared__ __align__(16) _Float16 Hs[ROWS_PER_BLOCK][HDIM];    // 32 KB

    const int tid   = threadIdx.x;
    const int wave  = tid >> 5;          // wave32
    const int lane  = tid & 31;
    const int wrow0 = wave * ROWS_PER_WAVE;
    const int rowBase = blockIdx.x * ROWS_PER_BLOCK + wrow0;

    // ---- stage A tile (16 rows x 128 K) into LDS, f16 ----
    for (int idx = lane; idx < ROWS_PER_WAVE * K1DIM; idx += 32) {
        int r = idx >> 7;
        int c = idx & 127;
        int grow = rowBase + r;
        int g = grow < ND ? grow : ND - 1;            // clamp tail (EXEC stays full)
        _Float16 v;
        if (c < FDIM) {
            int s = dispsec[g];
            v = hist16[(size_t)s * FDIM + c];
        } else {
            v = (_Float16)dispf[(size_t)g * FDIM + (c - FDIM)];
        }
        As[wrow0 + r][c] = v;
    }

    const _Float16* arow = &As[wrow0 + (lane & 15)][0];
    v16h a1[4];
#pragma unroll
    for (int ks = 0; ks < 4; ++ks) a1[ks] = load_afrag(arow, ks, lane);

    // ---- layer 1: 16 N-tiles, K = 128 ----
    for (int n0 = 0; n0 < HDIM; n0 += 16) {
        v8f acc = {};
#pragma unroll
        for (int ks = 0; ks < 4; ++ks) {
            v16h b = load_bfrag(bw1t, K1DIM, n0, ks, lane);
            acc = __builtin_amdgcn_wmma_f32_16x16x32_f16(
                false, a1[ks], false, b, (short)0, acc, false, false);
        }
        float bias = b1[n0 + (lane & 15)];
#pragma unroll
        for (int r = 0; r < 8; ++r) {
            float h = elu_fast(acc[r] + bias);
            int M = r + ((lane >> 4) << 3);             // C layout: VGPR r, lane hi-half -> M+8
            Hs[wrow0 + M][n0 + (lane & 15)] = (_Float16)h;
        }
    }

    // ---- layer 2 A fragments from Hs (K = 256) ----
    const _Float16* hrow = &Hs[wrow0 + (lane & 15)][0];
    v16h a2[8];
#pragma unroll
    for (int ks = 0; ks < 8; ++ks) a2[ks] = load_afrag(hrow, ks, lane);

    float part[8];
#pragma unroll
    for (int r = 0; r < 8; ++r) part[r] = 0.f;

    for (int n0 = 0; n0 < HDIM; n0 += 16) {
        v8f acc = {};
#pragma unroll
        for (int ks = 0; ks < 8; ++ks) {
            v16h b = load_bfrag(w2t, HDIM, n0, ks, lane);
            acc = __builtin_amdgcn_wmma_f32_16x16x32_f16(
                false, a2[ks], false, b, (short)0, acc, false, false);
        }
        float bias = b2[n0 + (lane & 15)];
        float w3v  = W3[n0 + (lane & 15)];
#pragma unroll
        for (int r = 0; r < 8; ++r) {
            float h = elu_fast(acc[r] + bias);
            part[r] += h * w3v;                         // layer-3 dot (per-lane N slice)
        }
    }

    // reduce over the 16 lanes that hold the N dimension (wave32 halves)
#pragma unroll
    for (int m = 1; m < 16; m <<= 1) {
#pragma unroll
        for (int r = 0; r < 8; ++r) part[r] += __shfl_xor(part[r], m, 32);
    }

    float b3v = b3p[0];
    if ((lane & 15) == 0) {
#pragma unroll
        for (int r = 0; r < 8; ++r) {
            int M = ((lane >> 4) << 3) + r;
            int grow = rowBase + M;
            if (grow < ND) {
                float uu = part[r] + b3v;
                u_out[grow]    = uu;
                expu_out[grow] = __expf(uu);
            }
        }
    }
}

// ---------------------------------------------------------------------------
// K3: per-session reductions. disp rows of session s are [s*per, s*per+per)
// (disp_2d_split_sec_ind is sorted with `per` rows/session in the reference).
// dense_exp row has <= per nonzeros -> top-2 from a tiny list (ties: lower item).
// ---------------------------------------------------------------------------
__global__ void session_kernel(const float* __restrict__ u,
                               const float* __restrict__ expu,
                               const int* __restrict__ clicksub,
                               const int* __restrict__ clickidx,   // [S,2]
                               const int* __restrict__ dispidx,    // [ND,2]
                               float* __restrict__ accum,
                               int S, int per) {
    int s = blockIdx.x * blockDim.x + threadIdx.x;
    if (s >= S) return;

    float se = 0.f;
    int   items[32];
    float vals[32];
    int   nu = 0;
    long long base = (long long)s * per;
    int lim = per < 32 ? per : 32;
    for (int j = 0; j < lim; ++j) {
        float e = expu[base + j];
        se += e;
        int it = dispidx[2 * (base + j) + 1];
        int k = 0;
        for (; k < nu; ++k) if (items[k] == it) { vals[k] += e; break; }
        if (k == nu) { items[nu] = it; vals[nu] = e; ++nu; }
    }

    // argmax / top-2 with jnp tie-break (lower item index wins)
    int bi = -1, si = -1;
    float bv = -1.f, sv = -1.f;
    for (int k = 0; k < nu; ++k) {
        float v = vals[k];
        int it = items[k];
        if (v > bv || (v == bv && it < bi)) {
            sv = bv; si = bi;
            bv = v;  bi = it;
        } else if (v > sv || (v == sv && it < si)) {
            sv = v;  si = it;
        }
    }
    if (nu < 2) si = (bi == 0) ? 1 : 0;   // zeros: lowest free index wins 2nd slot

    float uc = u[clicksub[s]];
    float lt = -uc + __logf(se + 1.0f);
    int ci = clickidx[2 * s + 1];
    float p1 = (ci == bi) ? 1.f : 0.f;
    float p2 = (ci == bi || ci == si) ? 1.f : 0.f;

    atomicAdd(&accum[0], lt);
    atomicAdd(&accum[1], p1);
    atomicAdd(&accum[2], p2);
    atomicAdd(&accum[3], 1.0f);           // event_cnt: one click per session
}

__global__ void zero_accum_kernel(float* accum) {
    if (threadIdx.x < 4) accum[threadIdx.x] = 0.f;
}

__global__ void finalize_kernel(const float* __restrict__ accum, float* __restrict__ out) {
    float ls = accum[0], p1 = accum[1], p2 = accum[2], ev = accum[3];
    out[0] = ls / ev;   // loss
    out[1] = p1 / ev;   // precision_1
    out[2] = p2 / ev;   // precision_2
    out[3] = ls;        // loss_sum
    out[4] = p1;        // p1_sum
    out[5] = p2;        // p2_sum
    out[6] = ev;        // event_cnt
}

// ---------------------------------------------------------------------------
// Host launcher
// ---------------------------------------------------------------------------
extern "C" void kernel_launch(void* const* d_in, const int* in_sizes, int n_in,
                              void* d_out, int out_size, void* d_ws, size_t ws_size,
                              hipStream_t stream) {
    const float* dispf    = (const float*)d_in[0];
    const float* Xs       = (const float*)d_in[1];
    const float* W1       = (const float*)d_in[2];
    const float* b1       = (const float*)d_in[3];
    const float* W2       = (const float*)d_in[4];
    const float* b2       = (const float*)d_in[5];
    const float* W3       = (const float*)d_in[6];
    const float* b3       = (const float*)d_in[7];
    const int*   tril     = (const int*)d_in[8];
    const int*   dispsec  = (const int*)d_in[10];
    const int*   clicksub = (const int*)d_in[11];
    const int*   clickidx = (const int*)d_in[12];
    const int*   dispidx  = (const int*)d_in[13];

    const int ND   = in_sizes[10];
    const int S    = in_sizes[11];
    const int F    = in_sizes[1] / S;          // 64
    const int H    = in_sizes[3];              // 256
    const int NT   = in_sizes[9];
    const int band = NT / S;
    const int per  = ND / S;
    const int PW   = in_sizes[2] / (F * H) - 1;
    (void)n_in; (void)out_size; (void)ws_size; (void)F; (void)H;

    char* w = (char*)d_ws;
    size_t off = 0;
    auto take = [&](size_t bytes) {
        char* p = w + off;
        off = (off + bytes + 255) & ~(size_t)255;
        return p;
    };
    _Float16* bw1t   = (_Float16*)take((size_t)HDIM * K1DIM * 2);
    _Float16* w2t    = (_Float16*)take((size_t)HDIM * HDIM * 2);
    _Float16* hist16 = (_Float16*)take((size_t)S * FDIM * 2);
    float*    u      = (float*)take((size_t)ND * 4);
    float*    expu   = (float*)take((size_t)ND * 4);
    float*    accum  = (float*)take(4 * 4);

    hipLaunchKernelGGL(prep_weights_kernel, dim3(384), dim3(256), 0, stream,
                       W1, W2, bw1t, w2t, PW);
    hipLaunchKernelGGL(hist_kernel, dim3(S), dim3(64), 0, stream,
                       Xs, tril, hist16, band);
    hipLaunchKernelGGL(zero_accum_kernel, dim3(1), dim3(64), 0, stream, accum);
    hipLaunchKernelGGL(mlp_fused_kernel, dim3((ND + ROWS_PER_BLOCK - 1) / ROWS_PER_BLOCK),
                       dim3(128), 0, stream,
                       dispf, dispsec, hist16, bw1t, w2t, b1, b2, W3, b3, u, expu, ND);
    hipLaunchKernelGGL(session_kernel, dim3((S + 255) / 256), dim3(256), 0, stream,
                       u, expu, clicksub, clickidx, dispidx, accum, S, per);
    hipLaunchKernelGGL(finalize_kernel, dim3(1), dim3(1), 0, stream, accum, (float*)d_out);
}
